// Decoder_498216206593
// MI455X (gfx1250) — matrix operand order repaired and checked by reading, not compiled, not run on hardware
//
#include <hip/hip_runtime.h>
#include <hip/hip_bf16.h>

typedef __attribute__((ext_vector_type(16))) _Float16 v16h;
typedef __attribute__((ext_vector_type(8)))  _Float16 v8h;
typedef __attribute__((ext_vector_type(8)))  float    v8f;

#define BROWS 1024
#define NL    64
#define NH    32
#define NGENE 20000
#define NC    64
#define GOI   1000
#define EPSV  1e-5f

// ---------------------------------------------------------------------------
// Kernel A: h = relu(BN(latent @ W + b)) -> f16  [1024][32] in workspace
// One wave (32 threads) per row; thread t computes column t (K=NL=64 dot).
// ---------------------------------------------------------------------------
__global__ __launch_bounds__(32) void mlp_head_kernel(
    const float* __restrict__ latent, const float* __restrict__ W,
    const float* __restrict__ bias,   const float* __restrict__ gamma,
    const float* __restrict__ beta,   const float* __restrict__ mean,
    const float* __restrict__ var,    _Float16* __restrict__ h_out)
{
    __shared__ float lat_s[NL];
    const int row = blockIdx.x;
    const int t   = threadIdx.x;            // 0..31
    lat_s[t]      = latent[row * NL + t];
    lat_s[t + 32] = latent[row * NL + t + 32];
    __syncthreads();
    float acc = 0.f;
#pragma unroll
    for (int l = 0; l < NL; ++l) acc = fmaf(lat_s[l], W[l * NH + t], acc);
    acc += bias[t];
    acc = (acc - mean[t]) * rsqrtf(var[t] + EPSV) * gamma[t] + beta[t];
    acc = fmaxf(acc, 0.f);
    h_out[row * NH + t] = (_Float16)acc;
}

// ---------------------------------------------------------------------------
// Shared GEMM core: one WMMA tile = 16x16 output, K=32 done in ONE
// v_wmma_f32_16x16x32_f16.  LDS holds Bs[K=32][N=64] (padded) as f16 and
// Hs[128 rows][K=32] (padded) as f16.
// ---------------------------------------------------------------------------
#define LDB 72   // 64 + 8 halfs pad
#define LDH 40   // 32 + 8 halfs pad

__device__ __forceinline__ v16h load_a_frag(const _Float16* Hs, int mt, int lane)
{
    const int rl   = mt * 16 + (lane & 15);
    const int koff = (lane < 16) ? 0 : 8;          // ISA 16-bit A layout
    v8h a0 = *(const v8h*)(&Hs[rl * LDH + koff]);        // K = koff..koff+7
    v8h a1 = *(const v8h*)(&Hs[rl * LDH + 16 + koff]);   // K = koff+16..koff+23
    return __builtin_shufflevector(a0, a1, 0,1,2,3,4,5,6,7,8,9,10,11,12,13,14,15);
}

__device__ __forceinline__ v16h load_b_frag(const _Float16* Bs, int nt, int lane)
{
    // B layout: lane%16 == N, elements over K in same order as A (koff..+7, 16+koff..+7)
    const int n    = nt * 16 + (lane & 15);
    const int koff = (lane < 16) ? 0 : 8;
    v16h b;
#pragma unroll
    for (int s = 0; s < 8; ++s) {
        b[s]     = Bs[(koff + s) * LDB + n];
        b[s + 8] = Bs[(16 + koff + s) * LDB + n];
    }
    return b;
}

__device__ __forceinline__ void stage_h(_Float16* Hs, const _Float16* __restrict__ h,
                                        int mbase, int tid)
{
    const int r   = tid >> 1;               // 0..127
    const int off = (tid & 1) * 16;         // 0 or 16 halfs
    v8h a = *(const v8h*)(h + (mbase + r) * NH + off);
    v8h b = *(const v8h*)(h + (mbase + r) * NH + off + 8);
    *(v8h*)(&Hs[r * LDH + off])     = a;
    *(v8h*)(&Hs[r * LDH + off + 8]) = b;
}

// ---------------------------------------------------------------------------
// Kernel B: logit[b,g,c] = sum_h h[b,h] * logit_table[genes_oi[g]][h][c]
// grid = (GOI, 8);  block = 256 (8 waves); each block: 1 gene x 128 rows.
// Output is write-once (256 MB > 192 MB L2) -> non-temporal stores so L2
// keeps the re-read gene weights and h instead.
// ---------------------------------------------------------------------------
__global__ __launch_bounds__(256) void logit_kernel(
    const _Float16* __restrict__ h,            // [1024][32] f16
    const float*    __restrict__ logit_table,  // [NG][32][64]
    const int*      __restrict__ genes_oi,     // [GOI]
    float*          __restrict__ out)          // [1024][GOI][64]
{
    __shared__ _Float16 Bs[NH * LDB];
    __shared__ _Float16 Hs[128 * LDH];

    const int g     = blockIdx.x;
    const int mbase = blockIdx.y * 128;
    const int tid   = threadIdx.x;

    // stage gathered gene weights [32][64] fp32 -> f16 LDS (already K-major)
    {
        const int gid = genes_oi[g];
        const float* lw = logit_table + (size_t)gid * (NH * NC);
        const int e = tid * 8;                 // 2048 elems / 256 thr
        const int k = e / NC, n = e % NC;
        float4 f0 = *(const float4*)(lw + e);
        float4 f1 = *(const float4*)(lw + e + 4);
        v8h v;
        v[0]=(_Float16)f0.x; v[1]=(_Float16)f0.y; v[2]=(_Float16)f0.z; v[3]=(_Float16)f0.w;
        v[4]=(_Float16)f1.x; v[5]=(_Float16)f1.y; v[6]=(_Float16)f1.z; v[7]=(_Float16)f1.w;
        *(v8h*)(&Bs[k * LDB + n]) = v;
    }
    stage_h(Hs, h, mbase, tid);
    __syncthreads();

    const int wave = tid >> 5, lane = tid & 31;
    const int nt = wave & 3;                   // N-tile 0..3
    const int m0 = wave >> 2;                  // 0..1

    const v16h bf = load_b_frag(Bs, nt, lane); // reused across 4 WMMAs

    const size_t ostride = (size_t)GOI * NC;
    const int    gcol    = g * NC + nt * 16 + (lane & 15);

#pragma unroll
    for (int i = 0; i < 4; ++i) {
        const int mt = m0 + i * 2;             // m-tiles m0, m0+2, m0+4, m0+6
        v16h af = load_a_frag(Hs, mt, lane);
        v8f c = {};
        c = __builtin_amdgcn_wmma_f32_16x16x32_f16(false, af, false, bf,
                                                   (short)0, c, false, false);
        const int grow = mbase + mt * 16 + ((lane < 16) ? 0 : 8);
#pragma unroll
        for (int r = 0; r < 8; ++r)
            __builtin_nontemporal_store(c[r], &out[(size_t)(grow + r) * ostride + gcol]);
    }
}

// ---------------------------------------------------------------------------
// Kernel C: rho[b,n] = sum_h h[b,h] * rho_table[n][h]
// grid = (ceil(NG/64), 8); block = 256.  Transpose staging: lane = K so the
// LDS write is a single ds_store_b128 per thread; global side is 8 dword
// loads, each fully coalesced across the wave (lane k consecutive).
// ---------------------------------------------------------------------------
__global__ __launch_bounds__(256) void rho_kernel(
    const _Float16* __restrict__ h,          // [1024][32] f16
    const float*    __restrict__ rho_table,  // [NG][32]
    float*          __restrict__ out)        // [1024][NG]
{
    __shared__ _Float16 Bs[NH * LDB];
    __shared__ _Float16 Hs[128 * LDH];

    const int nbase = blockIdx.x * 64;
    const int mbase = blockIdx.y * 128;
    const int tid   = threadIdx.x;

    // stage rho_table[nbase+j][k] -> Bs[k][j] (transpose, f16), zero-pad OOB
    {
        const int k  = tid & 31;               // lane = K row of Bs
        const int j0 = (tid >> 5) * 8;         // 8 consecutive N columns
        v8h v;
#pragma unroll
        for (int i = 0; i < 8; ++i) {
            const int n = nbase + j0 + i;
            float f = (n < NGENE) ? rho_table[(size_t)n * NH + k] : 0.f;
            v[i] = (_Float16)f;
        }
        *(v8h*)(&Bs[k * LDB + j0]) = v;        // one ds_store_b128
    }
    stage_h(Hs, h, mbase, tid);
    __syncthreads();

    const int wave = tid >> 5, lane = tid & 31;
    const int nt = wave & 3;
    const int m0 = wave >> 2;

    const v16h bf = load_b_frag(Bs, nt, lane);
    const int gcol = nbase + nt * 16 + (lane & 15);

#pragma unroll
    for (int i = 0; i < 4; ++i) {
        const int mt = m0 + i * 2;
        v16h af = load_a_frag(Hs, mt, lane);
        v8f c = {};
        c = __builtin_amdgcn_wmma_f32_16x16x32_f16(false, af, false, bf,
                                                   (short)0, c, false, false);
        if (gcol < NGENE) {
            const int grow = mbase + mt * 16 + ((lane < 16) ? 0 : 8);
#pragma unroll
            for (int r = 0; r < 8; ++r)
                __builtin_nontemporal_store(c[r], &out[(size_t)(grow + r) * NGENE + gcol]);
        }
    }
}

// ---------------------------------------------------------------------------
extern "C" void kernel_launch(void* const* d_in, const int* in_sizes, int n_in,
                              void* d_out, int out_size, void* d_ws, size_t ws_size,
                              hipStream_t stream)
{
    (void)in_sizes; (void)n_in; (void)out_size; (void)ws_size;
    const float* latent   = (const float*)d_in[0];
    const int*   genes_oi = (const int*)  d_in[1];
    const float* W        = (const float*)d_in[2];
    const float* bias     = (const float*)d_in[3];
    const float* gamma    = (const float*)d_in[4];
    const float* beta     = (const float*)d_in[5];
    const float* mean     = (const float*)d_in[6];
    const float* var      = (const float*)d_in[7];
    const float* ltable   = (const float*)d_in[8];
    const float* rtable   = (const float*)d_in[9];

    float* logit_out = (float*)d_out;                                  // [1024][1000][64]
    float* rho_out   = (float*)d_out + (size_t)BROWS * GOI * NC;       // [1024][20000]
    _Float16* h_f16  = (_Float16*)d_ws;                                // 64 KB

    mlp_head_kernel<<<BROWS, 32, 0, stream>>>(latent, W, bias, gamma, beta,
                                              mean, var, h_f16);
    logit_kernel<<<dim3(GOI, BROWS / 128), 256, 0, stream>>>(h_f16, ltable,
                                                             genes_oi, logit_out);
    rho_kernel<<<dim3((NGENE + 63) / 64, BROWS / 128), 256, 0, stream>>>(h_f16, rtable,
                                                                         rho_out);
}
